// BernMLPAugmenter_16724602651079
// MI455X (gfx1250) — matrix operand order repaired and checked by reading, not compiled, not run on hardware
//
#include <hip/hip_runtime.h>
#include <hip/hip_bf16.h>

typedef __attribute__((ext_vector_type(16))) _Float16 v16h;
typedef __attribute__((ext_vector_type(8)))  float    v8f;
typedef __attribute__((ext_vector_type(4)))  float    v4f;

#define NNODES 100000
#define DIM    128
#define EHALF  400000
#define HDIM   64
#define NTILES (EHALF / 16)      // 25000 tiles of 16 edges
#define WPB    8                 // waves per block
#define BLOCK  256
#define GRID   1250
#define BIASF  1e-4f

// ---------------------------------------------------------------------------
// prologue: zero the gate-sum accumulator (replay/graph safe)
// ---------------------------------------------------------------------------
__global__ void zero_ws_kernel(float* ws) {
    if (threadIdx.x == 0) ws[0] = 0.0f;
}

// ---------------------------------------------------------------------------
// epilogue: mean edge weight
// ---------------------------------------------------------------------------
__global__ void finalize_kernel(const float* ws, float* out) {
    if (threadIdx.x == 0) out[2 * EHALF] = ws[0] * (1.0f / (float)EHALF);
}

// ---------------------------------------------------------------------------
// main: per-wave 16-edge tile; 2x128 gather -> f16 WMMA 16x16x32 (K=256, N=64)
// -> in-register 64x1 second layer -> concrete gate -> symmetric outputs.
//
// __launch_bounds__(256, 4): >=4 waves/SIMD => <=256 VGPRs/wave.
// The W1 B-fragments live in LDS; an opaque (asm-laundered) zero offset is
// added to the LDS base INSIDE the tile loop so the compiler cannot hoist the
// 32 loop-invariant ds_load_b128s out of the loop (which previously caused
// either >256-VGPR allocation or scratch spill/reload around every WMMA).
// ---------------------------------------------------------------------------
__global__ __launch_bounds__(BLOCK, 4) void edge_mlp_wmma_kernel(
    const float* __restrict__ node_emb,   // [N, 128]
    const int*   __restrict__ src,        // [EH]
    const int*   __restrict__ dst,        // [EH]
    const float* __restrict__ vals,       // [EH]
    const float* __restrict__ eps_u,      // [EH]
    const float* __restrict__ W1,         // [256, 64]
    const float* __restrict__ b1,         // [64]
    const float* __restrict__ W2,         // [64]
    const float* __restrict__ b2,         // [1]
    float*       __restrict__ out,        // [2*EH (+1)]
    float*       __restrict__ gate_sum)   // [1] workspace
{
    // W1 staged as f16 B-fragments: sW1[(k*64 + n)*32 + kk] = W1[k*32+kk][n]
    __shared__ _Float16 sW1[8 * 64 * 32];   // 32 KB
    __shared__ float    sW2[64];

    const int tid = threadIdx.x;

    // cooperative, coalesced W1 f32->f16 stage (consecutive tid -> consecutive W1)
    for (int e = tid; e < 8 * 64 * 32; e += BLOCK) {
        const int n  = e & 63;
        const int kk = (e >> 6) & 31;
        const int k  = e >> 11;
        sW1[(k * 64 + n) * 32 + kk] = (_Float16)W1[(k * 32 + kk) * HDIM + n];
    }
    if (tid < HDIM) sW2[tid] = W2[tid];
    __syncthreads();

    const int lane = tid & 31;
    const int wave = tid >> 5;
    const int lo   = lane & 15;      // row / edge-in-tile this lane serves
    const int half = lane >> 4;      // 0: K-low lanes, 1: K-high lanes
    const int co   = half ? 8 : 0;   // unused; A fragment uses bo
    const int bo   = half ? 16 : 0;  // A/B-fragment K sub-offset (contiguous K order)

    // per-lane constants for the epilogue layer
    float b1v[4], w2v[4];
#pragma unroll
    for (int nt = 0; nt < 4; ++nt) {
        b1v[nt] = b1[nt * 16 + lo];
        w2v[nt] = sW2[nt * 16 + lo];
    }
    const float b2v = b2[0];

    const int waveGlobal = blockIdx.x * WPB + wave;
    const int nWaves     = GRID * WPB;

    for (int tile = waveGlobal; tile < NTILES; tile += nWaves) {
        const int ebase = tile * 16;
        // both half-wave groups need the same 16 row indices
        const long long srow = (long long)src[ebase + lo] * DIM;
        const long long drow = (long long)dst[ebase + lo] * DIM;
        const float* sp = node_emb + srow;
        const float* dp = node_emb + drow;

        // Opaque zero: forces the B-fragment LDS loads below to look
        // loop-variant so they are re-issued every tile (no LICM -> no
        // 256-VGPR hoist, no scratch spill).
        unsigned int zoff = 0;
        asm volatile("" : "+v"(zoff));
        const _Float16* w1p = sW1 + zoff;

        v8f acc[4] = {v8f{}, v8f{}, v8f{}, v8f{}};

#pragma unroll
        for (int k = 0; k < 8; ++k) {
            const float* rp = (k < 4) ? sp : dp;   // first 128 feats = src half
            const int kf = (k & 3) * 32;
            // A fragment (contiguous K order): lanes 0-15 K=0..15,
            // lanes 16-31 K=16..31 relative to kf
            v4f a0 = *(const v4f*)(rp + kf + bo);
            v4f a1 = *(const v4f*)(rp + kf + bo + 4);
            v4f a2 = *(const v4f*)(rp + kf + bo + 8);
            v4f a3 = *(const v4f*)(rp + kf + bo + 12);
            v16h a;
#pragma unroll
            for (int i = 0; i < 4; ++i) {
                a[i]      = (_Float16)a0[i];
                a[4 + i]  = (_Float16)a1[i];
                a[8 + i]  = (_Float16)a2[i];
                a[12 + i] = (_Float16)a3[i];
            }
#pragma unroll
            for (int nt = 0; nt < 4; ++nt) {
                const int n = nt * 16 + lo;
                // B fragment (contiguous K order, as A): lanes 0-15 K=0..15, 16-31 K=16..31
                const v16h b = *(const v16h*)&w1p[(k * 64 + n) * 32 + bo];
                acc[nt] = __builtin_amdgcn_wmma_f32_16x16x32_f16(
                    false, a, false, b, (short)0, acc[nt], false, false);
            }
        }

        // ---- second layer in registers: p[j] = sum_nt relu(h)*w2 ----------
        // lane holds h[j + 8*half][nt*16 + lo] in acc[nt][j]
        float p[8];
#pragma unroll
        for (int j = 0; j < 8; ++j) {
            float s = 0.0f;
#pragma unroll
            for (int nt = 0; nt < 4; ++nt) {
                float h = acc[nt][j] + b1v[nt];
                h = h > 0.0f ? h : 0.0f;
                s += h * w2v[nt];
            }
            // reduce the 16 lanes of this half (columns 0..63 covered)
            s += __shfl_xor(s, 1, 32);
            s += __shfl_xor(s, 2, 32);
            s += __shfl_xor(s, 4, 32);
            s += __shfl_xor(s, 8, 32);
            p[j] = s;   // lanes 0-15: logit row j; lanes 16-31: logit row j+8
        }

        // distribute row logits: lane with lo==m (both halves) gets edge m
        float logit = 0.0f;
#pragma unroll
        for (int j = 0; j < 8; ++j) {
            const float lowv = __shfl(p[j], 0, 32);    // row j
            const float hiv  = __shfl(p[j], 16, 32);   // row j+8
            if (lo == j)     logit = lowv;
            if (lo == j + 8) logit = hiv;
        }
        logit += b2v;

        // ---- concrete / gumbel-sigmoid gate -------------------------------
        const float u   = eps_u[ebase + lo];
        const float eps = (2.0f * BIASF - 1.0f) * u + (1.0f - BIASF);
        const float gi  = __logf(eps) - __logf(1.0f - eps) + logit;
        const float gate = 1.0f / (1.0f + __expf(-gi));
        const float nv   = vals[ebase + lo] * gate;

        if (half == 0) out[ebase + lo] = nv;            // new_vals
        else           out[EHALF + ebase + lo] = nv;    // symmetric duplicate

        // tile gate sum (lanes 0-15 distinct; upper half duplicates)
        float gs = gate;
        gs += __shfl_xor(gs, 1, 32);
        gs += __shfl_xor(gs, 2, 32);
        gs += __shfl_xor(gs, 4, 32);
        gs += __shfl_xor(gs, 8, 32);
        if (lane == 0) atomicAdd(gate_sum, gs);
    }
}

extern "C" void kernel_launch(void* const* d_in, const int* in_sizes, int n_in,
                              void* d_out, int out_size, void* d_ws, size_t ws_size,
                              hipStream_t stream) {
    (void)in_sizes; (void)n_in; (void)out_size; (void)ws_size;
    const float* node_emb = (const float*)d_in[0];
    const int*   src      = (const int*)d_in[1];
    const int*   dst      = (const int*)d_in[2];
    const float* vals     = (const float*)d_in[3];
    const float* eps_u    = (const float*)d_in[4];
    const float* W1       = (const float*)d_in[5];
    const float* b1       = (const float*)d_in[6];
    const float* W2       = (const float*)d_in[7];
    const float* b2       = (const float*)d_in[8];
    float* out = (float*)d_out;
    float* ws  = (float*)d_ws;

    zero_ws_kernel<<<1, 32, 0, stream>>>(ws);
    edge_mlp_wmma_kernel<<<GRID, BLOCK, 0, stream>>>(
        node_emb, src, dst, vals, eps_u, W1, b1, W2, b2, out, ws);
    finalize_kernel<<<1, 32, 0, stream>>>(ws, out);
}
